// LocalAttention_1005022347801
// MI455X (gfx1250) — matrix operand order repaired and checked by reading, not compiled, not run on hardware
//
#include <hip/hip_runtime.h>
#include <hip/hip_bf16.h>

typedef __attribute__((ext_vector_type(16))) __bf16 v16bf;
typedef __attribute__((ext_vector_type(8)))  __bf16 v8bf;
typedef __attribute__((ext_vector_type(8)))  float  v8f;
typedef __attribute__((ext_vector_type(4)))  float  f4;
typedef __attribute__((ext_vector_type(4)))  int    v4i;

#define BATCH 4
#define TLEN  2048
#define HDIM  1024
#define KSZ   16
#define TPAD  2080   // 7 left pad + 2048 + 8 right pad, rounded up to 32-row multiple

#if defined(__HIP_DEVICE_COMPILE__) && __has_builtin(__builtin_amdgcn_global_load_async_to_lds_b128)
#define HAVE_ASYNC_LDS 1
#else
#define HAVE_ASYNC_LDS 0
#endif

// ---------------------------------------------------------------------------
// Build zero-padded bf16 hi/lo split of inputs: xp[b][p][h], p in [0,TPAD)
// valid data at p in [7, 7+TLEN); zeros elsewhere.
// ---------------------------------------------------------------------------
__global__ __launch_bounds__(256) void prep_xpad(const float* __restrict__ x,
                                                 __bf16* __restrict__ hi,
                                                 __bf16* __restrict__ lo) {
  long i = (long)blockIdx.x * blockDim.x + threadIdx.x;
  const long total = (long)BATCH * TPAD * HDIM;
  if (i >= total) return;
  int h = (int)(i % HDIM);
  long r = i / HDIM;
  int p = (int)(r % TPAD);
  int b = (int)(r / TPAD);
  float val = 0.0f;
  if (p >= 7 && p < 7 + TLEN)
    val = x[((long)b * TLEN + (p - 7)) * HDIM + h];
  __bf16 hv = (__bf16)val;
  hi[i] = hv;
  lo[i] = (__bf16)(val - (float)hv);
}

// ---------------------------------------------------------------------------
// Transposed bf16 hi/lo split of a weight: wT[n][k] = w[k][n]
// ---------------------------------------------------------------------------
__global__ __launch_bounds__(256) void prep_wT(const float* __restrict__ w,
                                               __bf16* __restrict__ hi,
                                               __bf16* __restrict__ lo) {
  long i = (long)blockIdx.x * blockDim.x + threadIdx.x;
  const long total = (long)HDIM * HDIM;
  if (i >= total) return;
  int k = (int)(i % HDIM);
  int n = (int)(i / HDIM);
  float val = w[(long)k * HDIM + n];
  __bf16 hv = (__bf16)val;
  hi[i] = hv;
  lo[i] = (__bf16)(val - (float)hv);
}

// ---------------------------------------------------------------------------
// Fragment loaders.
//   A fragment (16x32 bf16): lane<16 -> M=lane,    K = 0..15 (contiguous)
//                            lane>=16 -> M=lane-16, K = 16..31 (contiguous)
//   B fragment (32x16 bf16): lane<16 -> N=lane,    K = 0..15 (contiguous)
//                            lane>=16 -> N=lane-16, K = 16..31 (contiguous)
// ---------------------------------------------------------------------------
__device__ __forceinline__ v16bf load_afrag(const __bf16* p) {
  v8bf x0 = *(const v8bf*)(p);
  v8bf x1 = *(const v8bf*)(p + 8);
  return __builtin_shufflevector(x0, x1, 0,1,2,3,4,5,6,7,8,9,10,11,12,13,14,15);
}
__device__ __forceinline__ v16bf load_bfrag(const __bf16* p) {
  return *(const v16bf*)(p);
}
#define WMMA_BF16(A, B, C) \
  __builtin_amdgcn_wmma_f32_16x16x32_bf16(false, (A), false, (B), (short)0, (C), false, false)

// ---------------------------------------------------------------------------
// 2x2 register-blocked wave GEMM: each wave computes a 32x32 C tile,
// C[m,n] = sum_k A[m,k] * W[k,n] with transposed W (Bmat[n][k]) and
// 3-term bf16 split (hi*hi + hi*lo + lo*hi) for ~fp32 accuracy.
// 12 WMMAs per 32-K step reuse each A/B fragment twice -> 2x less cache
// traffic per WMMA than a wave-per-16x16 scheme.
// ---------------------------------------------------------------------------
__global__ __launch_bounds__(256) void gemm_bf16split(
    const __bf16* __restrict__ Ahi, const __bf16* __restrict__ Alo,
    const __bf16* __restrict__ Bhi, const __bf16* __restrict__ Blo,
    float* __restrict__ C,
    int tiles32M, int rowOff, long strideA, long strideC, int totalWaves) {
  int wave = blockIdx.x * (blockDim.x >> 5) + (threadIdx.x >> 5);
  if (wave >= totalWaves) return;              // wave-uniform branch
  int lane = threadIdx.x & 31;

  const int tiles32N = HDIM / 32;              // 32
  int perB = tiles32M * tiles32N;
  int b   = wave / perB;
  int rem = wave % perB;
  int tm  = rem / tiles32N;
  int tn  = rem % tiles32N;

  long arow0 = (long)b * strideA + (long)(rowOff + tm * 32 + (lane & 15)) * HDIM;
  long arow1 = arow0 + 16L * HDIM;
  long brow0 = (long)(tn * 32 + (lane & 15)) * HDIM;
  long brow1 = brow0 + 16L * HDIM;
  int koff = (lane & 16) ? 16 : 0;             // A-fragment K sub-offset
  int kb   = (lane & 16) ? 16 : 0;             // B-fragment K sub-offset

  v8f c00 = {}, c01 = {}, c10 = {}, c11 = {};
#pragma unroll 4
  for (int kk = 0; kk < HDIM; kk += 32) {
    const __bf16* pa0h = Ahi + arow0 + kk + koff;
    const __bf16* pa0l = Alo + arow0 + kk + koff;
    const __bf16* pa1h = Ahi + arow1 + kk + koff;
    const __bf16* pa1l = Alo + arow1 + kk + koff;
    v16bf a0h = load_afrag(pa0h);
    v16bf a0l = load_afrag(pa0l);
    v16bf a1h = load_afrag(pa1h);
    v16bf a1l = load_afrag(pa1l);
    v16bf b0h = load_bfrag(Bhi + brow0 + kk + kb);
    v16bf b0l = load_bfrag(Blo + brow0 + kk + kb);
    v16bf b1h = load_bfrag(Bhi + brow1 + kk + kb);
    v16bf b1l = load_bfrag(Blo + brow1 + kk + kb);

    if (kk + 32 < HDIM) {                      // prefetch next K slice
      __builtin_prefetch(pa0h + 32, 0, 3);
      __builtin_prefetch(pa1h + 32, 0, 3);
      __builtin_prefetch(Bhi + brow0 + kk + kb + 32, 0, 3);
      __builtin_prefetch(Bhi + brow1 + kk + kb + 32, 0, 3);
    }

    c00 = WMMA_BF16(a0h, b0h, c00);
    c00 = WMMA_BF16(a0h, b0l, c00);
    c00 = WMMA_BF16(a0l, b0h, c00);
    c01 = WMMA_BF16(a0h, b1h, c01);
    c01 = WMMA_BF16(a0h, b1l, c01);
    c01 = WMMA_BF16(a0l, b1h, c01);
    c10 = WMMA_BF16(a1h, b0h, c10);
    c10 = WMMA_BF16(a1h, b0l, c10);
    c10 = WMMA_BF16(a1l, b0h, c10);
    c11 = WMMA_BF16(a1h, b1h, c11);
    c11 = WMMA_BF16(a1h, b1l, c11);
    c11 = WMMA_BF16(a1l, b1h, c11);
  }

  // D layout: VGPR r -> row (lane<16 ? r : r+8), col = n0 + (lane&15)
  long cbase = (long)b * strideC + (long)(tm * 32) * HDIM + tn * 32;
  int col = lane & 15;
  int r0  = (lane & 16) ? 8 : 0;
#pragma unroll
  for (int r = 0; r < 8; ++r) {
    C[cbase + (long)(r0 + r) * HDIM + col]            = c00[r];
    C[cbase + (long)(r0 + r) * HDIM + col + 16]       = c01[r];
    C[cbase + (long)(16 + r0 + r) * HDIM + col]       = c10[r];
    C[cbase + (long)(16 + r0 + r) * HDIM + col + 16]  = c11[r];
  }
}

// ---------------------------------------------------------------------------
// Per-(b,t) block: score -> softmax -> weighted sum.  16 waves, wave k owns
// score_k = sum_h tanh(proj[b,t+k,h] + q[b,t,h]) * v[h].
// The 16 x_pad rows needed for the output are staged into LDS with
// global_load_async_to_lds_b128 issued BEFORE the tanh phase; the wave only
// waits on ASYNCcnt after its score reduction -> copy latency fully hidden.
// ---------------------------------------------------------------------------
__global__ __launch_bounds__(512) void attn_kernel(
    const float* __restrict__ x, const float* __restrict__ q,
    const float* __restrict__ proj, const float* __restrict__ v,
    float* __restrict__ out) {
  __shared__ float sq[HDIM];
  __shared__ float sv[HDIM];
  __shared__ float sx[KSZ * HDIM];             // 16 x_pad rows (64 KB)
  __shared__ float ss[KSZ];
  __shared__ float sw[KSZ];

  int bt = blockIdx.x;
  int b = bt / TLEN, t = bt % TLEN;
  int tid = threadIdx.x;
  int w = tid >> 5, lane = tid & 31;

  long qbase = ((long)b * TLEN + t) * HDIM;
  sq[tid]       = q[qbase + tid];
  sq[tid + 512] = q[qbase + tid + 512];
  sv[tid]       = v[tid];
  sv[tid + 512] = v[tid + 512];

  // ---- stage x_pad row w (k == w) into LDS: 8 chunks of 512B per wave ----
  int roww = t + w - 7;                        // unpadded input row for k = w
  char* dstBase = (char*)&sx[(size_t)w * HDIM];
  if (roww >= 0 && roww < TLEN) {
    char* srcBase = (char*)(x + ((long)b * TLEN + roww) * HDIM);
#if HAVE_ASYNC_LDS
#pragma unroll
    for (int i = 0; i < 8; ++i) {
      __builtin_amdgcn_global_load_async_to_lds_b128(
          (__attribute__((address_space(1))) v4i*)(srcBase + lane * 16 + i * 512),
          (__attribute__((address_space(3))) v4i*)(dstBase + lane * 16 + i * 512),
          0, 0);
    }
#else
    const f4* s4 = (const f4*)srcBase;
    f4* d4 = (f4*)dstBase;
#pragma unroll
    for (int i = 0; i < 8; ++i) d4[lane + i * 32] = s4[lane + i * 32];
#endif
  } else {
    f4 z = {0.f, 0.f, 0.f, 0.f};
    f4* d4 = (f4*)dstBase;
#pragma unroll
    for (int i = 0; i < 8; ++i) d4[lane + i * 32] = z;
  }

  __syncthreads();                             // sq/sv visible

  // ---- score_k for k == w (overlaps with the async x copies) ----
  const float* pr = proj + ((long)b * TPAD + (t + w)) * HDIM;
  float acc = 0.0f;
#pragma unroll 4
  for (int i = 0; i < 32; ++i) {
    int h = lane + i * 32;
    acc += tanhf(pr[h] + sq[h]) * sv[h];
  }
#pragma unroll
  for (int m = 16; m >= 1; m >>= 1) acc += __shfl_xor(acc, m, 32);
  if (lane == 0) ss[w] = acc;

#if HAVE_ASYNC_LDS
  asm volatile("s_wait_asynccnt 0x0" ::: "memory");   // per-wave async copies done
#endif
  __syncthreads();                             // ss + sx visible

  if (tid < KSZ) {
    float mx = ss[0];
#pragma unroll
    for (int j = 1; j < KSZ; ++j) mx = fmaxf(mx, ss[j]);
    sw[tid] = __expf(ss[tid] - mx);
  }
  __syncthreads();

  float den = 0.0f;
#pragma unroll
  for (int j = 0; j < KSZ; ++j) den += sw[j];
  float inv = 1.0f / den;

#pragma unroll
  for (int hh = 0; hh < 2; ++hh) {
    int h = tid + hh * 512;
    float o = 0.0f;
#pragma unroll
    for (int k = 0; k < KSZ; ++k)
      o += sw[k] * sx[k * HDIM + h];           // rows pre-zeroed when invalid
    out[((long)b * TLEN + t) * HDIM + h] = o * inv;
  }
}

// ---------------------------------------------------------------------------
extern "C" void kernel_launch(void* const* d_in, const int* in_sizes, int n_in,
                              void* d_out, int out_size, void* d_ws, size_t ws_size,
                              hipStream_t stream) {
  const float* x  = (const float*)d_in[0];
  const float* w1 = (const float*)d_in[1];
  const float* w2 = (const float*)d_in[2];
  const float* v  = (const float*)d_in[3];
  float* out = (float*)d_out;

  char* ws = (char*)d_ws;
  size_t off = 0;
  auto take = [&](size_t bytes) -> char* {
    char* p = ws + off;
    off = (off + bytes + 255) & ~(size_t)255;
    return p;
  };

  const long XPE = (long)BATCH * TPAD * HDIM;     // padded input elements
  const long WE  = (long)HDIM * HDIM;

  __bf16* xp_hi  = (__bf16*)take(XPE * 2);
  __bf16* xp_lo  = (__bf16*)take(XPE * 2);
  __bf16* w1T_hi = (__bf16*)take(WE * 2);
  __bf16* w1T_lo = (__bf16*)take(WE * 2);
  __bf16* w2T_hi = (__bf16*)take(WE * 2);
  __bf16* w2T_lo = (__bf16*)take(WE * 2);
  float*  qbuf   = (float*)take((long)BATCH * TLEN * HDIM * 4);
  float*  pbuf   = (float*)take((long)BATCH * TPAD * HDIM * 4);

  // 1) padded bf16 split of inputs
  prep_xpad<<<(unsigned)((XPE + 255) / 256), 256, 0, stream>>>(x, xp_hi, xp_lo);
  // 2) transposed bf16 splits of weights
  prep_wT<<<(unsigned)((WE + 255) / 256), 256, 0, stream>>>(w1, w1T_hi, w1T_lo);
  prep_wT<<<(unsigned)((WE + 255) / 256), 256, 0, stream>>>(w2, w2T_hi, w2T_lo);
  // 3) q = inputs @ w1 : 32-row tiles start at padded row 7
  {
    int tiles32M = TLEN / 32;                     // 64
    int total    = BATCH * tiles32M * (HDIM / 32);// 8192 wave-tiles
    int blocks   = (total + 7) / 8;               // 8 waves per 256-thread block
    gemm_bf16split<<<blocks, 256, 0, stream>>>(
        xp_hi, xp_lo, w1T_hi, w1T_lo, qbuf,
        tiles32M, /*rowOff=*/7,
        (long)TPAD * HDIM, (long)TLEN * HDIM, total);
  }
  // 4) proj = x_pad @ w2 : all TPAD rows
  {
    int tiles32M = TPAD / 32;                     // 65
    int total    = BATCH * tiles32M * (HDIM / 32);// 8320 wave-tiles
    int blocks   = (total + 7) / 8;
    gemm_bf16split<<<blocks, 256, 0, stream>>>(
        xp_hi, xp_lo, w2T_hi, w2T_lo, pbuf,
        tiles32M, /*rowOff=*/0,
        (long)TPAD * HDIM, (long)TPAD * HDIM, total);
  }
  // 5) score -> softmax -> weighted sum
  attn_kernel<<<BATCH * TLEN, 512, 0, stream>>>(x, qbuf, pbuf, v, out);
}
